// MultiHeadRelativeAttention_16930761080990
// MI455X (gfx1250) — matrix operand order repaired and checked by reading, not compiled, not run on hardware
//
#include <hip/hip_runtime.h>
#include <hip/hip_bf16.h>
#include <math.h>
#include <stdint.h>

typedef __bf16 bf16_t;
typedef __attribute__((ext_vector_type(16))) __bf16 v16bf;
typedef __attribute__((ext_vector_type(8)))  float  v8f;

#define QL     2048
#define KLN    2048
#define NB     2
#define EMB    1024
#define NH     16
#define HD     64
#define MQB    4096          /* QL*NB rows for the big GEMMs */
#define KPROWS 2304          /* 2048 + 256 zero padding      */
#define KPPAD  128

union FragAB { v16bf v; unsigned u[8]; };
union FragC  { v8f   v; float    f[8]; };

// 16-bit A-matrix 16x32 (MxK) per-lane K index (ISA 7.12.2 table)
__device__ __forceinline__ int afrag_k(int j, int lane) {
  int kb = (lane >> 4) << 3;
  return (j < 4) ? (kb + 2 * j) : (16 + kb + 2 * (j - 4));
}
// 16-bit B-matrix 32x16 (KxN) per-lane K index
__device__ __forceinline__ int bfrag_k(int j, int lane) {
  int kb = (lane >> 4) << 3;
  return (j < 4) ? (kb + 2 * j) : (16 + kb + 2 * (j - 4));
}

// ---------------------------------------------------------------------------
// Tiled WMMA GEMM:  C[M,N](f32) = A[M,K](bf16, row-major) * W[N,K](bf16)^T
// (i.e. torch Linear: x @ W.T with W stored [out,in] exactly as given)
// Block tile 128x128, 8 waves (2x4), each wave 64x32, K-step 32.
// K-slabs stream into double-buffered LDS via GLOBAL_LOAD_ASYNC_TO_LDS_B128
// (ASYNCcnt path): next slab loads while WMMAs consume the current one.
// ---------------------------------------------------------------------------
__global__ __launch_bounds__(256)
void gemm_bf16_wmma(const bf16_t* __restrict__ A, const bf16_t* __restrict__ W,
                    float* __restrict__ C, int M, int N, int K)
{
  __shared__ bf16_t Als[2][128][32];
  __shared__ bf16_t Bls[2][128][32];
  const int t = threadIdx.x, lane = t & 31, wave = t >> 5;
  const int mW = wave & 1, nW = wave >> 1;
  const int mBase = blockIdx.y * 128, nBase = blockIdx.x * 128;
  const int cn = lane & 15, hl = lane >> 4;

  FragC acc[4][2];
  #pragma unroll
  for (int i = 0; i < 4; ++i)
    #pragma unroll
    for (int j = 0; j < 2; ++j)
      #pragma unroll
      for (int v = 0; v < 8; ++v) acc[i][j].f[v] = 0.f;

  // 256 thr * 2 chunks * 8 bf16 (16B) = one 128x32 tile per matrix
  auto prefetch = [&](int kk, int buf) {
    #pragma unroll
    for (int p = 0; p < 2; ++p) {
      int e = (t + p * 256) * 8;
      int r = e >> 5, c = e & 31;
      unsigned la = (unsigned)(uintptr_t)&Als[buf][r][c];   // flat->LDS: low 32 bits
      unsigned lb = (unsigned)(uintptr_t)&Bls[buf][r][c];
      const bf16_t* ga = &A[(size_t)(mBase + r) * K + kk + c];
      const bf16_t* gb = &W[(size_t)(nBase + r) * K + kk + c];
      asm volatile("global_load_async_to_lds_b128 %0, %1, off"
                   :: "v"(la), "v"(ga) : "memory");
      asm volatile("global_load_async_to_lds_b128 %0, %1, off"
                   :: "v"(lb), "v"(gb) : "memory");
    }
  };

  prefetch(0, 0);                       // 4 async ops outstanding
  for (int kk = 0; kk < K; kk += 32) {
    const int buf = (kk >> 5) & 1;
    const bool more = (kk + 32) < K;
    if (more) {
      prefetch(kk + 32, buf ^ 1);       // overlap next slab with this compute
      // async loads retire in order: <=4 outstanding => current slab landed
      asm volatile("s_wait_asynccnt 4" ::: "memory");
    } else {
      asm volatile("s_wait_asynccnt 0" ::: "memory");
    }
    __syncthreads();                    // all waves' slabs visible

    FragAB a[4], b[2];
    #pragma unroll
    for (int i = 0; i < 4; ++i) {
      int row = mW * 64 + i * 16 + cn;
      #pragma unroll
      for (int j = 0; j < 8; ++j)
        a[i].u[j] = *(const unsigned*)&Als[buf][row][afrag_k(j, lane)];
    }
    #pragma unroll
    for (int jb = 0; jb < 2; ++jb) {
      int row = nW * 32 + jb * 16 + cn;
      #pragma unroll
      for (int j = 0; j < 8; ++j)
        b[jb].u[j] = *(const unsigned*)&Bls[buf][row][bfrag_k(j, lane)];
    }
    #pragma unroll
    for (int i = 0; i < 4; ++i)
      #pragma unroll
      for (int jb = 0; jb < 2; ++jb)
        acc[i][jb].v = __builtin_amdgcn_wmma_f32_16x16x32_bf16(
            false, a[i].v, false, b[jb].v, (short)0, acc[i][jb].v, false, false);

    asm volatile("s_wait_dscnt 0" ::: "memory");  // our LDS reads done
    __syncthreads();                    // buffer free for overwrite next round
  }

  #pragma unroll
  for (int i = 0; i < 4; ++i)
    #pragma unroll
    for (int jb = 0; jb < 2; ++jb)
      #pragma unroll
      for (int v = 0; v < 8; ++v) {
        int m = mBase + mW * 64 + i * 16 + v + hl * 8;
        int n = nBase + nW * 32 + jb * 16 + cn;
        C[(size_t)m * N + n] = acc[i][jb].f[v];
      }
}

// ---------------------------------------------------------------------------
// Fused flash attention with Transformer-XL relative positions.
// Grid (QL/128, NH, NB), 256 threads = 8 waves, each wave owns 16 q-rows.
// pos-shift handled via a 16x32 Toeplitz window GEMM + LDS diagonal gather:
//   pos[q,col] = Qp[q] . Kpos2[q-col],  Kpos2 padded by KPPAD zero rows.
// ---------------------------------------------------------------------------
__global__ __launch_bounds__(256)
void attn_relpos_flash(const bf16_t* __restrict__ Qc, const bf16_t* __restrict__ Qp,
                       const bf16_t* __restrict__ Kh, const bf16_t* __restrict__ Vt,
                       const bf16_t* __restrict__ Kp2, bf16_t* __restrict__ Obf)
{
  __shared__ float  Tlds[8][16][32];
  __shared__ bf16_t Plds[8][16][32];

  const int t = threadIdx.x, lane = t & 31, wave = t >> 5;
  const int cn = lane & 15, hl = lane >> 4;
  const int b = blockIdx.z, h = blockIdx.y;
  const int q0blk = blockIdx.x * 128;
  const int q0w   = q0blk + wave * 16;

  const size_t bh = (size_t)(b * NH + h);
  const bf16_t* Qcb = Qc + bh * (size_t)QL  * HD;
  const bf16_t* Qpb = Qp + bh * (size_t)QL  * HD;
  const bf16_t* Khb = Kh + bh * (size_t)KLN * HD;
  const bf16_t* Vtb = Vt + bh * (size_t)HD  * KLN;   // [d][k] transposed
  const bf16_t* Kpb = Kp2 + (size_t)h * KPROWS * HD;

  // Q fragments (A-layout), two 32-wide d chunks, kept resident
  FragAB qcf[2], qpf[2];
  #pragma unroll
  for (int dc = 0; dc < 2; ++dc)
    #pragma unroll
    for (int j = 0; j < 8; ++j) {
      int k = dc * 32 + afrag_k(j, lane);
      qcf[dc].u[j] = *(const unsigned*)&Qcb[(size_t)(q0w + cn) * HD + k];
      qpf[dc].u[j] = *(const unsigned*)&Qpb[(size_t)(q0w + cn) * HD + k];
    }

  FragC o[4];
  #pragma unroll
  for (int dfr = 0; dfr < 4; ++dfr)
    #pragma unroll
    for (int v = 0; v < 8; ++v) o[dfr].f[v] = 0.f;
  float mrun[8], lrun[8];
  #pragma unroll
  for (int v = 0; v < 8; ++v) { mrun[v] = -INFINITY; lrun[v] = 0.f; }

  const int nChunk = (q0blk + 128) >> 5;          // causal: cols <= block end
  for (int ch = 0; ch < nChunk; ++ch) {
    const int c0 = ch << 5;
    if (c0 + 64 < KLN)                             // prefetch next K tile
      __builtin_prefetch(&Khb[(size_t)(c0 + 64 + cn) * HD], 0, 1);

    float sv[2][8];
    #pragma unroll
    for (int sub = 0; sub < 2; ++sub) {
      const int c0s = c0 + (sub << 4);
      // content scores: S = Qc (16x64) x K^T tile (64x16)
      FragC s;
      #pragma unroll
      for (int v = 0; v < 8; ++v) s.f[v] = 0.f;
      #pragma unroll
      for (int dc = 0; dc < 2; ++dc) {
        FragAB kb;
        #pragma unroll
        for (int j = 0; j < 8; ++j)
          kb.u[j] = *(const unsigned*)&Khb[(size_t)(c0s + cn) * HD + dc * 32 + bfrag_k(j, lane)];
        s.v = __builtin_amdgcn_wmma_f32_16x16x32_bf16(
            false, qcf[dc].v, false, kb.v, (short)0, s.v, false, false);
      }
      // positional Toeplitz window: T[r,tt] = Qp[r] . Kpos2[rowbase+tt]
      const int rowbase = q0w - c0s - 15 + KPPAD;  // always >= 1
      #pragma unroll
      for (int tb = 0; tb < 2; ++tb) {
        FragC tw;
        #pragma unroll
        for (int v = 0; v < 8; ++v) tw.f[v] = 0.f;
        #pragma unroll
        for (int dc = 0; dc < 2; ++dc) {
          FragAB pb;
          #pragma unroll
          for (int j = 0; j < 8; ++j)
            pb.u[j] = *(const unsigned*)&Kpb[(size_t)(rowbase + tb * 16 + cn) * HD + dc * 32 + bfrag_k(j, lane)];
          tw.v = __builtin_amdgcn_wmma_f32_16x16x32_bf16(
              false, qpf[dc].v, false, pb.v, (short)0, tw.v, false, false);
        }
        #pragma unroll
        for (int v = 0; v < 8; ++v)
          Tlds[wave][v + hl * 8][tb * 16 + cn] = tw.f[v];
      }
      asm volatile("s_wait_dscnt 0" ::: "memory");
      // diagonal gather (relative shift), scale, causal mask
      #pragma unroll
      for (int v = 0; v < 8; ++v) {
        int r = v + hl * 8;
        float val = (s.f[v] + Tlds[wave][r][r - cn + 15]) * 0.125f;
        sv[sub][v] = ((c0s + cn) > (q0w + r)) ? -INFINITY : val;
      }
    }

    // online softmax over the 32-col chunk (row = 16 lanes of one half-wave)
    float fr[8];
    #pragma unroll
    for (int v = 0; v < 8; ++v) {
      float mx = fmaxf(sv[0][v], sv[1][v]);
      mx = fmaxf(mx, __shfl_xor(mx, 1, 32));
      mx = fmaxf(mx, __shfl_xor(mx, 2, 32));
      mx = fmaxf(mx, __shfl_xor(mx, 4, 32));
      mx = fmaxf(mx, __shfl_xor(mx, 8, 32));
      float mnew = fmaxf(mrun[v], mx);
      float fv = __expf(mrun[v] - mnew);
      float p0 = __expf(sv[0][v] - mnew);
      float p1 = __expf(sv[1][v] - mnew);
      float rs = p0 + p1;
      rs += __shfl_xor(rs, 1, 32);
      rs += __shfl_xor(rs, 2, 32);
      rs += __shfl_xor(rs, 4, 32);
      rs += __shfl_xor(rs, 8, 32);
      lrun[v] = lrun[v] * fv + rs;
      mrun[v] = mnew;
      fr[v] = fv;
      Plds[wave][v + hl * 8][cn]      = (bf16_t)p0;
      Plds[wave][v + hl * 8][16 + cn] = (bf16_t)p1;
    }
    #pragma unroll
    for (int dfr = 0; dfr < 4; ++dfr)
      #pragma unroll
      for (int v = 0; v < 8; ++v) o[dfr].f[v] *= fr[v];

    asm volatile("s_wait_dscnt 0" ::: "memory");
    // PV: O(16x64) += P(16x32) x V(32x64)
    FragAB pa;
    #pragma unroll
    for (int j = 0; j < 8; ++j)
      pa.u[j] = *(const unsigned*)&Plds[wave][cn][afrag_k(j, lane)];
    #pragma unroll
    for (int dfr = 0; dfr < 4; ++dfr) {
      FragAB vb;
      #pragma unroll
      for (int j = 0; j < 8; ++j)
        vb.u[j] = *(const unsigned*)&Vtb[(size_t)(dfr * 16 + cn) * KLN + c0 + bfrag_k(j, lane)];
      o[dfr].v = __builtin_amdgcn_wmma_f32_16x16x32_bf16(
          false, pa.v, false, vb.v, (short)0, o[dfr].v, false, false);
    }
  }

  // epilogue: normalize and store as [q, b, h*64+d] bf16 for the output GEMM
  #pragma unroll
  for (int dfr = 0; dfr < 4; ++dfr)
    #pragma unroll
    for (int v = 0; v < 8; ++v) {
      int r = v + hl * 8;
      int q = q0w + r;
      float val = o[dfr].f[v] / lrun[v];
      Obf[(size_t)(q * NB + b) * EMB + h * HD + dfr * 16 + cn] = (bf16_t)val;
    }
}

// ---------------------------------------------------------------------------
// Conversion / packing kernels
// ---------------------------------------------------------------------------
__global__ void k_f32_to_bf16(const float* __restrict__ s, bf16_t* __restrict__ d, int n) {
  int i = blockIdx.x * 256 + threadIdx.x;
  if (i < n) d[i] = (bf16_t)s[i];
}

// pe_slice = pe[1 : 1+4095] padded to 4096 rows with zeros
__global__ void k_pe_slice_bf16(const float* __restrict__ pe, bf16_t* __restrict__ d) {
  int i = blockIdx.x * 256 + threadIdx.x;      // 4096*1024
  int row = i >> 10, col = i & 1023;
  float v = (row < 2 * KLN - 1) ? pe[(size_t)(row + 1) * EMB + col] : 0.f;
  d[i] = (bf16_t)v;
}

__global__ void k_pack_q(const float* __restrict__ Cq, const float* __restrict__ cb,
                         const float* __restrict__ pb,
                         bf16_t* __restrict__ Qc, bf16_t* __restrict__ Qp) {
  int i = blockIdx.x * 256 + threadIdx.x;      // (q*2+b)*1024 + h*64 + d
  int q = i >> 11, b = (i >> 10) & 1, hd = i & 1023;
  int h = hd >> 6, dd = hd & 63;
  float v = Cq[i];
  size_t dst = ((size_t)(b * NH + h) * QL + q) * HD + dd;
  Qc[dst] = (bf16_t)(v + cb[hd]);
  Qp[dst] = (bf16_t)(v + pb[hd]);
}

__global__ void k_pack_k(const float* __restrict__ Ck, bf16_t* __restrict__ Kh) {
  int i = blockIdx.x * 256 + threadIdx.x;
  int c = i >> 11, b = (i >> 10) & 1, hd = i & 1023;
  int h = hd >> 6, dd = hd & 63;
  Kh[((size_t)(b * NH + h) * KLN + c) * HD + dd] = (bf16_t)Ck[i];
}

__global__ void k_pack_v(const float* __restrict__ Cv, bf16_t* __restrict__ Vt) {
  int i = blockIdx.x * 256 + threadIdx.x;
  int c = i >> 11, b = (i >> 10) & 1, hd = i & 1023;
  int h = hd >> 6, dd = hd & 63;
  Vt[((size_t)(b * NH + h) * HD + dd) * KLN + c] = (bf16_t)Cv[i];   // transposed
}

// Kp2[h][t+KPPAD][d] = k_pos[h][2047-t][d] for t in [0,2047], zeros elsewhere
__global__ void k_pack_kpos(const float* __restrict__ Ckp, bf16_t* __restrict__ Kp2) {
  int i = blockIdx.x * 256 + threadIdx.x;      // NH*KPROWS*HD
  if (i >= NH * KPROWS * HD) return;
  int h = i / (KPROWS * HD);
  int rem = i - h * (KPROWS * HD);
  int tp = rem / HD, dd = rem % HD;
  float v = 0.f;
  if (tp >= KPPAD && tp <= KPPAD + KLN - 1) {
    int s = (KLN - 1) - (tp - KPPAD);
    v = Ckp[(size_t)s * EMB + h * HD + dd];
  }
  Kp2[i] = (bf16_t)v;
}

// ---------------------------------------------------------------------------
extern "C" void kernel_launch(void* const* d_in, const int* in_sizes, int n_in,
                              void* d_out, int out_size, void* d_ws, size_t ws_size,
                              hipStream_t stream)
{
  (void)in_sizes; (void)n_in; (void)out_size; (void)ws_size;
  const float* q_in  = (const float*)d_in[0];
  const float* k_in  = (const float*)d_in[1];
  const float* v_in  = (const float*)d_in[2];
  const float* pe    = (const float*)d_in[3];
  const float* w_q   = (const float*)d_in[4];
  const float* w_k   = (const float*)d_in[5];
  const float* w_v   = (const float*)d_in[6];
  const float* w_kp  = (const float*)d_in[7];
  const float* w_o   = (const float*)d_in[8];
  const float* cbias = (const float*)d_in[9];
  const float* pbias = (const float*)d_in[10];
  // d_in[11] (attn_mask) ignored: causal mask applied analytically.

  char* wsb = (char*)d_ws;
  size_t off = 0;
  auto alloc = [&](size_t bytes) -> void* {
    void* p = wsb + off; off += (bytes + 255) & ~(size_t)255; return p;
  };
  const size_t SZ_ACT = (size_t)MQB * EMB * sizeof(bf16_t);  // 8 MB
  const size_t SZ_W   = (size_t)EMB * EMB * sizeof(bf16_t);  // 2 MB
  const size_t SZ_C   = (size_t)MQB * EMB * sizeof(float);   // 16 MB
  const size_t SZ_BH  = (size_t)NB * NH * QL * HD * sizeof(bf16_t);

  bf16_t* Aq  = (bf16_t*)alloc(SZ_ACT);
  bf16_t* Ak  = (bf16_t*)alloc(SZ_ACT);
  bf16_t* Av  = (bf16_t*)alloc(SZ_ACT);
  bf16_t* Ape = (bf16_t*)alloc(SZ_ACT);
  bf16_t* Wq  = (bf16_t*)alloc(SZ_W);
  bf16_t* Wk  = (bf16_t*)alloc(SZ_W);
  bf16_t* Wv  = (bf16_t*)alloc(SZ_W);
  bf16_t* Wkp = (bf16_t*)alloc(SZ_W);
  bf16_t* Wo  = (bf16_t*)alloc(SZ_W);
  float*  Cq  = (float*)alloc(SZ_C);
  float*  Ck  = (float*)alloc(SZ_C);
  float*  Cv  = (float*)alloc(SZ_C);
  float*  Ckp = (float*)alloc(SZ_C);
  bf16_t* Qc  = (bf16_t*)alloc(SZ_BH);
  bf16_t* Qp  = (bf16_t*)alloc(SZ_BH);
  bf16_t* Kh  = (bf16_t*)alloc(SZ_BH);
  bf16_t* Vt  = (bf16_t*)alloc(SZ_BH);
  bf16_t* Kp2 = (bf16_t*)alloc((size_t)NH * KPROWS * HD * sizeof(bf16_t));
  bf16_t* Obf = (bf16_t*)alloc(SZ_ACT);

  const int nAct = MQB * EMB;      // 4194304
  const int nW   = EMB * EMB;      // 1048576
  dim3 blk(256);

  k_f32_to_bf16<<<nAct / 256, blk, 0, stream>>>(q_in, Aq, nAct);
  k_f32_to_bf16<<<nAct / 256, blk, 0, stream>>>(k_in, Ak, nAct);
  k_f32_to_bf16<<<nAct / 256, blk, 0, stream>>>(v_in, Av, nAct);
  k_pe_slice_bf16<<<nAct / 256, blk, 0, stream>>>(pe, Ape);
  k_f32_to_bf16<<<nW / 256, blk, 0, stream>>>(w_q,  Wq,  nW);
  k_f32_to_bf16<<<nW / 256, blk, 0, stream>>>(w_k,  Wk,  nW);
  k_f32_to_bf16<<<nW / 256, blk, 0, stream>>>(w_v,  Wv,  nW);
  k_f32_to_bf16<<<nW / 256, blk, 0, stream>>>(w_kp, Wkp, nW);
  k_f32_to_bf16<<<nW / 256, blk, 0, stream>>>(w_o,  Wo,  nW);

  dim3 gg(EMB / 128, MQB / 128);   // (8, 32)
  gemm_bf16_wmma<<<gg, blk, 0, stream>>>(Aq,  Wq,  Cq,  MQB, EMB, EMB);
  gemm_bf16_wmma<<<gg, blk, 0, stream>>>(Ak,  Wk,  Ck,  MQB, EMB, EMB);
  gemm_bf16_wmma<<<gg, blk, 0, stream>>>(Av,  Wv,  Cv,  MQB, EMB, EMB);
  gemm_bf16_wmma<<<gg, blk, 0, stream>>>(Ape, Wkp, Ckp, MQB, EMB, EMB);

  k_pack_q<<<nAct / 256, blk, 0, stream>>>(Cq, cbias, pbias, Qc, Qp);
  k_pack_k<<<nAct / 256, blk, 0, stream>>>(Ck, Kh);
  k_pack_v<<<nAct / 256, blk, 0, stream>>>(Cv, Vt);
  const int nKp = NH * KPROWS * HD;
  k_pack_kpos<<<(nKp + 255) / 256, blk, 0, stream>>>(Ckp, Kp2);

  dim3 ga(QL / 128, NH, NB);       // (16, 16, 2)
  attn_relpos_flash<<<ga, blk, 0, stream>>>(Qc, Qp, Kh, Vt, Kp2, Obf);

  gemm_bf16_wmma<<<gg, blk, 0, stream>>>(Obf, Wo, (float*)d_out, MQB, EMB, EMB);
}
